// MulitheadAttention_1649267441965
// MI455X (gfx1250) — compile-verified, matrix operand order checked by reading, not run on hardware
//
#include <hip/hip_runtime.h>
#include <hip/hip_bf16.h>

// ---------------- problem constants ----------------
#define NHEAD 16
#define DMODEL 1024
#define DK 64
#define DV 64
#define BB 2
#define SS 2048
#define MROWS (BB * SS)          // 4096 flattened (b,s) rows
#define LN_EPS 1e-5f
#define INV_TEMP 0.125f          // 1/sqrt(64)

// fused attention kernel strip: 16 query rows per workgroup
#define STRIP 16
#define LDP 2052                 // padded LDS row stride (floats): distinct banks

// ---------------- WMMA types & helpers ----------------
typedef __attribute__((ext_vector_type(16))) _Float16 v16h;
typedef __attribute__((ext_vector_type(8)))  _Float16 v8h;
typedef __attribute__((ext_vector_type(8)))  float    v8f;

union FragH {
    v16h v;
    v8h  p[2];
    _Float16 h[16];
};

__device__ __forceinline__ v8f wmma_f16(v16h a, v16h b, v8f c) {
    // D = A(16x32 f16) * B(32x16 f16) + C(16x16 f32)
    return __builtin_amdgcn_wmma_f32_16x16x32_f16(
        false, a, false, b, (short)0, c, false, false);
}

// Load a 16x32 A-fragment (or 32x16 B-fragment via K-contiguous rows of the
// [N,K] matrix) from row-major f16 memory. ISA layout:
//   lanes 0-15 : row = lane,    K chunks {0..7, 16..23}
//   lanes 16-31: row = lane-16, K chunks {8..15, 24..31}
__device__ __forceinline__ v16h load_frag_h(const _Float16* __restrict__ tile,
                                            int ld, int lane) {
    const int r = lane & 15, hi = lane >> 4;
    const _Float16* p = tile + (size_t)r * ld + hi * 8;
    FragH f;
    f.p[0] = *(const v8h*)(p);
    f.p[1] = *(const v8h*)(p + 16);
    return f.v;
}

// Same fragment built from f32 LDS (converted in-flight) — attn @ V stage.
__device__ __forceinline__ v16h load_frag_lds(const float* sc, int kc, int lane) {
    const int r = lane & 15, hi = lane >> 4;
    const float* p = sc + r * LDP + kc + hi * 8;
    FragH f;
#pragma unroll
    for (int i = 0; i < 8; ++i) {
        f.h[i]     = (_Float16)p[i];
        f.h[i + 8] = (_Float16)p[i + 16];
    }
    return f.v;
}

// ---------------- kernel 1: f32 -> f16 ----------------
__global__ void cvt_f16_kernel(const float* __restrict__ src,
                               _Float16* __restrict__ dst, int n) {
    for (int i = blockIdx.x * blockDim.x + threadIdx.x; i < n;
         i += gridDim.x * blockDim.x)
        dst[i] = (_Float16)src[i];
}

// ---------------- kernel 2: QKV projection GEMM ----------------
// out = A[4096,1024] @ W[1024,1024]^T + bias ; written f16 to
//   transposed==0 : [B, H, S, Dk]
//   transposed==1 : [B, H, Dv, S]   (so attn@V B-fragments are contiguous)
__global__ void qkv_proj_kernel(const _Float16* __restrict__ A,
                                const _Float16* __restrict__ W,
                                const float* __restrict__ bias,
                                _Float16* __restrict__ out, int transposed) {
    const int lane = threadIdx.x & 31;
    const int w = blockIdx.x * (blockDim.x >> 5) + (threadIdx.x >> 5);
    const int tm = w & 127;          // 128 M-tiles of 32
    const int tn = w >> 7;           // 32 N-tiles of 32
    const int m0 = tm * 32, n0 = tn * 32;

    v8f c00 = {}, c01 = {}, c10 = {}, c11 = {};
    for (int kk = 0; kk < DMODEL; kk += 32) {
        v16h a0 = load_frag_h(A + (size_t)m0 * DMODEL + kk, DMODEL, lane);
        v16h a1 = load_frag_h(A + (size_t)(m0 + 16) * DMODEL + kk, DMODEL, lane);
        v16h b0 = load_frag_h(W + (size_t)n0 * DMODEL + kk, DMODEL, lane);
        v16h b1 = load_frag_h(W + (size_t)(n0 + 16) * DMODEL + kk, DMODEL, lane);
        c00 = wmma_f16(a0, b0, c00);
        c01 = wmma_f16(a0, b1, c01);
        c10 = wmma_f16(a1, b0, c10);
        c11 = wmma_f16(a1, b1, c11);
    }

    const int r = lane & 15, hi = lane >> 4;
#pragma unroll
    for (int st = 0; st < 4; ++st) {
        const int sm = (st >> 1) * 16, sn = (st & 1) * 16;
        const v8f c = (st == 0) ? c00 : (st == 1) ? c01 : (st == 2) ? c10 : c11;
        const int col = n0 + sn + r;
        const int h = col >> 6, d = col & 63;
        const float bv = bias[col];
#pragma unroll
        for (int i = 0; i < 8; ++i) {
            const int row = m0 + sm + i + hi * 8;
            const int b = row >> 11, s = row & (SS - 1);
            const float val = c[i] + bv;
            size_t idx = transposed
                ? (((size_t)(b * NHEAD + h) * DV + d) * SS + s)
                : (((size_t)(b * NHEAD + h) * SS + s) * DK + d);
            out[idx] = (_Float16)val;
        }
    }
}

// ---------------- kernel 3: fused QK^T -> softmax -> attn@V ----------------
// One workgroup (8 waves) owns a strip of 16 query rows for one (b,h):
//   stage 1: scores strip (16 x 2048 f32) -> LDS via WMMA
//   stage 2: row softmax in LDS (wave32 shuffle reductions); write attn once
//   stage 3: ctx strip = attn_strip @ V, K split across waves, ds_add_f32
//            combine in LDS, single coalesced f16 ctx store.
__global__ void attn_fused_kernel(const _Float16* __restrict__ qp,
                                  const _Float16* __restrict__ kp,
                                  const _Float16* __restrict__ vpT,
                                  float* __restrict__ attn,
                                  _Float16* __restrict__ ctx) {
    __shared__ float sc[STRIP * LDP];      // 131,328 B score strip
    __shared__ float cacc[STRIP * DV];     //   4,096 B ctx accumulator

    const int z = blockIdx.z;              // b*H + h
    const int row0 = blockIdx.x * STRIP;   // query-row strip origin
    const int wave = threadIdx.x >> 5;
    const int lane = threadIdx.x & 31;
    const int r = lane & 15, hi = lane >> 4;

    const _Float16* Qb = qp + (size_t)z * SS * DK;
    const _Float16* Kb = kp + (size_t)z * SS * DK;
    const _Float16* Vb = vpT + (size_t)z * DV * SS;  // [DV, SS]

    // zero ctx accumulator (covered by the stage-1 barrier)
    for (int e = threadIdx.x; e < STRIP * DV; e += 256) cacc[e] = 0.f;

    // ---- stage 1: scores = (Q strip) @ K^T * inv_temp -> LDS ----
    const v16h aq0 = load_frag_h(Qb + (size_t)row0 * DK + 0,  DK, lane);
    const v16h aq1 = load_frag_h(Qb + (size_t)row0 * DK + 32, DK, lane);
    for (int ct = wave; ct < SS / 32; ct += 8) {
        const int n0 = ct * 32;
        v8f c0 = {}, c1 = {};
        v16h b0 = load_frag_h(Kb + (size_t)n0 * DK + 0, DK, lane);
        v16h b1 = load_frag_h(Kb + (size_t)(n0 + 16) * DK + 0, DK, lane);
        c0 = wmma_f16(aq0, b0, c0);
        c1 = wmma_f16(aq0, b1, c1);
        b0 = load_frag_h(Kb + (size_t)n0 * DK + 32, DK, lane);
        b1 = load_frag_h(Kb + (size_t)(n0 + 16) * DK + 32, DK, lane);
        c0 = wmma_f16(aq1, b0, c0);
        c1 = wmma_f16(aq1, b1, c1);
#pragma unroll
        for (int i = 0; i < 8; ++i) {
            sc[(i + hi * 8) * LDP + n0 + r]      = c0[i] * INV_TEMP;
            sc[(i + hi * 8) * LDP + n0 + 16 + r] = c1[i] * INV_TEMP;
        }
    }
    __syncthreads();

    // ---- stage 2: stable softmax per row; write attn (only pass) ----
    for (int rr = wave * 2; rr < wave * 2 + 2; ++rr) {
        float* row = sc + rr * LDP;
        float m = -3.4e38f;
        for (int j = lane; j < SS; j += 32) m = fmaxf(m, row[j]);
#pragma unroll
        for (int off = 16; off > 0; off >>= 1)
            m = fmaxf(m, __shfl_xor(m, off, 32));
        float sum = 0.f;
        for (int j = lane; j < SS; j += 32) {
            const float e = __expf(row[j] - m);
            row[j] = e;
            sum += e;
        }
#pragma unroll
        for (int off = 16; off > 0; off >>= 1)
            sum += __shfl_xor(sum, off, 32);
        const float inv = 1.0f / sum;
        float* gout = attn + (size_t)z * SS * SS + (size_t)(row0 + rr) * SS;
        for (int j = lane; j < SS; j += 32) {
            const float p = row[j] * inv;
            row[j] = p;          // keep for stage 3
            gout[j] = p;         // the one mandatory attn write
        }
    }
    __syncthreads();

    // ---- stage 3: ctx strip = attn_strip @ V ; K split across waves ----
    v8f d0 = {}, d1 = {}, d2 = {}, d3 = {};
    for (int kc = wave * (SS / 8); kc < (wave + 1) * (SS / 8); kc += 32) {
        const v16h a = load_frag_lds(sc, kc, lane);
        const v16h b0 = load_frag_h(Vb + (size_t)0  * SS + kc, SS, lane);
        const v16h b1 = load_frag_h(Vb + (size_t)16 * SS + kc, SS, lane);
        const v16h b2 = load_frag_h(Vb + (size_t)32 * SS + kc, SS, lane);
        const v16h b3 = load_frag_h(Vb + (size_t)48 * SS + kc, SS, lane);
        d0 = wmma_f16(a, b0, d0);
        d1 = wmma_f16(a, b1, d1);
        d2 = wmma_f16(a, b2, d2);
        d3 = wmma_f16(a, b3, d3);
    }
#pragma unroll
    for (int t = 0; t < 4; ++t) {
        const v8f d = (t == 0) ? d0 : (t == 1) ? d1 : (t == 2) ? d2 : d3;
        const int c0 = t * 16 + r;
#pragma unroll
        for (int i = 0; i < 8; ++i)
            atomicAdd(&cacc[(i + hi * 8) * DV + c0], d[i]);
    }
    __syncthreads();

    // write ctx strip (f16) to [B*S, H*DV]
    const int b = z >> 4, h = z & 15;
    for (int e = threadIdx.x; e < STRIP * DV; e += 256) {
        const int rr = e >> 6, cc = e & 63;
        const size_t g = (size_t)(b * SS + row0 + rr);
        ctx[g * DMODEL + h * DV + cc] = (_Float16)cacc[e];
    }
}

// ---------------- kernel 4: out-proj + bias + residual ----------------
__global__ void fproj_kernel(const _Float16* __restrict__ ctx,
                             const _Float16* __restrict__ Wf,
                             const float* __restrict__ bf,
                             const float* __restrict__ residual,
                             float* __restrict__ pre) {
    const int lane = threadIdx.x & 31;
    const int w = blockIdx.x * (blockDim.x >> 5) + (threadIdx.x >> 5);
    const int tm = w & 127, tn = w >> 7;
    const int m0 = tm * 32, n0 = tn * 32;

    v8f c00 = {}, c01 = {}, c10 = {}, c11 = {};
    for (int kk = 0; kk < DMODEL; kk += 32) {
        v16h a0 = load_frag_h(ctx + (size_t)m0 * DMODEL + kk, DMODEL, lane);
        v16h a1 = load_frag_h(ctx + (size_t)(m0 + 16) * DMODEL + kk, DMODEL, lane);
        v16h b0 = load_frag_h(Wf + (size_t)n0 * DMODEL + kk, DMODEL, lane);
        v16h b1 = load_frag_h(Wf + (size_t)(n0 + 16) * DMODEL + kk, DMODEL, lane);
        c00 = wmma_f16(a0, b0, c00);
        c01 = wmma_f16(a0, b1, c01);
        c10 = wmma_f16(a1, b0, c10);
        c11 = wmma_f16(a1, b1, c11);
    }

    const int r = lane & 15, hi = lane >> 4;
#pragma unroll
    for (int st = 0; st < 4; ++st) {
        const int sm = (st >> 1) * 16, sn = (st & 1) * 16;
        const v8f c = (st == 0) ? c00 : (st == 1) ? c01 : (st == 2) ? c10 : c11;
        const int col = n0 + sn + r;
        const float bv = bf[col];
#pragma unroll
        for (int i = 0; i < 8; ++i) {
            const int row = m0 + sm + i + hi * 8;
            const size_t idx = (size_t)row * DMODEL + col;
            pre[idx] = c[i] + bv + residual[idx];
        }
    }
}

// ---------------- kernel 5: LayerNorm ----------------
__global__ void ln_kernel(const float* __restrict__ pre,
                          const float* __restrict__ gamma,
                          const float* __restrict__ beta,
                          float* __restrict__ out) {
    const size_t row = blockIdx.x;
    const float* p = pre + row * DMODEL;
    __shared__ float rs[256];
    __shared__ float rs2[256];
    const int t = threadIdx.x;

    float v[4];
    float s = 0.f, s2 = 0.f;
#pragma unroll
    for (int i = 0; i < 4; ++i) {
        v[i] = p[t + i * 256];
        s += v[i];
        s2 += v[i] * v[i];
    }
    rs[t] = s;
    rs2[t] = s2;
    __syncthreads();
    for (int sft = 128; sft > 0; sft >>= 1) {
        if (t < sft) {
            rs[t] += rs[t + sft];
            rs2[t] += rs2[t + sft];
        }
        __syncthreads();
    }
    const float mu = rs[0] * (1.0f / DMODEL);
    const float var = rs2[0] * (1.0f / DMODEL) - mu * mu;
    const float inv = rsqrtf(var + LN_EPS);
#pragma unroll
    for (int i = 0; i < 4; ++i) {
        const int c = t + i * 256;
        out[row * DMODEL + c] = (v[i] - mu) * inv * gamma[c] + beta[c];
    }
}

// ---------------- launcher ----------------
extern "C" void kernel_launch(void* const* d_in, const int* in_sizes, int n_in,
                              void* d_out, int out_size, void* d_ws, size_t ws_size,
                              hipStream_t stream) {
    (void)in_sizes; (void)n_in; (void)out_size; (void)ws_size;

    const float* q     = (const float*)d_in[0];
    const float* k     = (const float*)d_in[1];
    const float* v     = (const float*)d_in[2];
    const float* Wq    = (const float*)d_in[3];
    const float* bq    = (const float*)d_in[4];
    const float* Wk    = (const float*)d_in[5];
    const float* bk    = (const float*)d_in[6];
    const float* Wf    = (const float*)d_in[7];
    const float* bf    = (const float*)d_in[8];
    const float* gamma = (const float*)d_in[9];
    const float* beta  = (const float*)d_in[10];

    float* out  = (float*)d_out;
    float* attn = out + (size_t)MROWS * DMODEL;   // [B,H,S,S]

    // workspace layout (bytes); buffers reused once their producers are done
    char* ws = (char*)d_ws;
    const size_t SZ_ACT_H = (size_t)MROWS * DMODEL * sizeof(_Float16);   // 8 MB
    const size_t SZ_W_H   = (size_t)DMODEL * DMODEL * sizeof(_Float16);  // 2 MB
    _Float16* q_h   = (_Float16*)(ws + 0 * SZ_ACT_H);
    _Float16* k_h   = (_Float16*)(ws + 1 * SZ_ACT_H);
    _Float16* v_h   = (_Float16*)(ws + 2 * SZ_ACT_H);
    _Float16* Wq_h  = (_Float16*)(ws + 3 * SZ_ACT_H);
    _Float16* Wk_h  = (_Float16*)(ws + 3 * SZ_ACT_H + 1 * SZ_W_H);
    _Float16* Wf_h  = (_Float16*)(ws + 3 * SZ_ACT_H + 2 * SZ_W_H);
    _Float16* qp_h  = (_Float16*)(ws + 3 * SZ_ACT_H + 3 * SZ_W_H);
    _Float16* kp_h  = (_Float16*)(ws + 4 * SZ_ACT_H + 3 * SZ_W_H);
    _Float16* vpT_h = (_Float16*)(ws + 5 * SZ_ACT_H + 3 * SZ_W_H);
    // reuse: ctx over q_h (free after q-proj), preLN over k_h+v_h (16 MB)
    _Float16* ctx_h = q_h;
    float*    pre   = (float*)k_h;

    const int NACT = MROWS * DMODEL;   // 4,194,304
    const int NW   = DMODEL * DMODEL;  // 1,048,576

    // 1) f32 -> f16 conversions
    cvt_f16_kernel<<<4096, 256, 0, stream>>>(q,  q_h,  NACT);
    cvt_f16_kernel<<<4096, 256, 0, stream>>>(k,  k_h,  NACT);
    cvt_f16_kernel<<<4096, 256, 0, stream>>>(v,  v_h,  NACT);
    cvt_f16_kernel<<<2048, 256, 0, stream>>>(Wq, Wq_h, NW);
    cvt_f16_kernel<<<2048, 256, 0, stream>>>(Wk, Wk_h, NW);
    cvt_f16_kernel<<<2048, 256, 0, stream>>>(Wf, Wf_h, NW);

    // 2) projections (note: V uses Wk/bk, faithful to the reference)
    qkv_proj_kernel<<<512, 256, 0, stream>>>(q_h, Wq_h, bq, qp_h, 0);
    qkv_proj_kernel<<<512, 256, 0, stream>>>(k_h, Wk_h, bk, kp_h, 0);
    qkv_proj_kernel<<<512, 256, 0, stream>>>(v_h, Wk_h, bk, vpT_h, 1);

    // 3) fused QK^T -> softmax -> attn@V (attn written exactly once)
    attn_fused_kernel<<<dim3(SS / STRIP, 1, BB * NHEAD), 256, 0, stream>>>(
        qp_h, kp_h, vpT_h, attn, ctx_h);

    // 4) output projection + bias + residual
    fproj_kernel<<<512, 256, 0, stream>>>(ctx_h, Wf_h, bf, q, pre);

    // 5) LayerNorm -> d_out
    ln_kernel<<<MROWS, 256, 0, stream>>>(pre, gamma, beta, out);
}
